// ContextualAttention_34763465293903
// MI455X (gfx1250) — compile-verified, matrix operand order checked by reading, not run on hardware
//
#include <hip/hip_runtime.h>

// ---------------------------------------------------------------------------
// ContextualAttention for MI455X (gfx1250, wave32, WMMA bf16 16x16x32 + TDM).
// Shapes fixed by the reference: B=2, S=2048, C=2048, T=4096, E=1024, H=16, D=64.
// ---------------------------------------------------------------------------

typedef __attribute__((ext_vector_type(16))) __bf16 v16bf;
typedef __attribute__((ext_vector_type(8)))  float  v8f;
typedef __attribute__((ext_vector_type(4)))  unsigned int v4u;
typedef __attribute__((ext_vector_type(4)))  int    v4i;
typedef __attribute__((ext_vector_type(8)))  int    v8i;

union FragBF16 {            // one WMMA bf16 A/B fragment: 16 bf16 = 8 VGPRs
  v16bf v;
  uint4 u[2];
};

#define EMB       1024
#define HEADS     16
#define HEADSIZE  64
#define BATCH     2
#define SEQ       2048
#define CTX       2048
#define TOT       (SEQ + CTX)
#define INV_SCALE 0.1767766952966369f   // 1 / 1024^0.25

#if defined(__gfx1250__) && __has_builtin(__builtin_amdgcn_tensor_load_to_lds)
#define HAS_TDM 1
#else
#define HAS_TDM 0
#endif

// ---- bf16 <-> f32 helpers (round-to-nearest-even) --------------------------
__device__ __forceinline__ unsigned short f2bf(float f) {
  union { float f; unsigned u; } x; x.f = f;
  unsigned r = x.u + 0x7fffu + ((x.u >> 16) & 1u);
  return (unsigned short)(r >> 16);
}
__device__ __forceinline__ float bf2f(unsigned short h) {
  union { unsigned u; float f; } x; x.u = ((unsigned)h) << 16;
  return x.f;
}

// ---- cross-lane reductions (wave32) ----------------------------------------
__device__ __forceinline__ float redmax16(float v) {   // within each 16-lane half
  for (int o = 8; o >= 1; o >>= 1) v = fmaxf(v, __shfl_xor(v, o, 32));
  return v;
}
__device__ __forceinline__ float redsum16(float v) {
  for (int o = 8; o >= 1; o >>= 1) v += __shfl_xor(v, o, 32);
  return v;
}

#if HAS_TDM
// ---------------------------------------------------------------------------
// TDM: DMA a 2D tile (tile_d0 elems/row x tile_d1 rows, 2-byte elements,
// row stride stride0 elems) from global memory into LDS at lds_byte_addr.
// D# layout per CDNA5 ISA 8.3/8.4: group0 = {flags, lds_addr, gaddr lo/hi|type},
// group1 = {data_size, dims, tile dims, strides}. Groups 2/3 unused (2D).
// ---------------------------------------------------------------------------
__device__ __forceinline__ void tdm_load_tile_2d(unsigned lds_byte_addr,
                                                 unsigned long long gaddr,
                                                 unsigned tile_d0,
                                                 unsigned tile_d1,
                                                 unsigned long long stride0) {
  v4u g0;
  g0[0] = 1u;                                        // count=1 (valid), user mode
  g0[1] = lds_byte_addr;                             // lds_addr (bytes)
  g0[2] = (unsigned)(gaddr & 0xFFFFFFFFull);         // global_addr[31:0]
  g0[3] = (unsigned)((gaddr >> 32) & 0x01FFFFFFull)  // global_addr[56:32]
          | (2u << 30);                              // type = 2 ("image")

  const unsigned td0 = 1u << 20;                     // large tensor dims: no OOB clip
  const unsigned td1 = 1u << 20;
  v8i g1;
  g1[0] = (int)(1u << 16);                           // data_size=1 (2B); mask/flags=0
  g1[1] = (int)((td0 & 0xFFFFu) << 16);              // tensor_dim0[15:0] @ bits 63:48
  g1[2] = (int)(((td0 >> 16) & 0xFFFFu) | ((td1 & 0xFFFFu) << 16));
  g1[3] = (int)(((td1 >> 16) & 0xFFFFu) | ((tile_d0 & 0xFFFFu) << 16)); // tile_dim0
  g1[4] = (int)(tile_d1 & 0xFFFFu);                  // tile_dim1; tile_dim2=0
  g1[5] = (int)(unsigned)(stride0 & 0xFFFFFFFFull);  // tensor_dim0_stride[31:0]
  g1[6] = (int)(unsigned)((stride0 >> 32) & 0xFFFFull); // stride0[47:32]; stride1 lo=0
  g1[7] = 0;                                         // stride1 hi (unused, 2D)

  v4i z4 = {0, 0, 0, 0};
#if __clang_major__ >= 23
  v8i z8 = {0, 0, 0, 0, 0, 0, 0, 0};
  __builtin_amdgcn_tensor_load_to_lds(g0, g1, z4, z4, z8, 0);
#else
  __builtin_amdgcn_tensor_load_to_lds(g0, g1, z4, z4, 0);
#endif
}
#endif  // HAS_TDM

// ---------------------------------------------------------------------------
// 1) f32 rows -> bf16 rows with row remapping (builds x_bf16 and xc_bf16).
// ---------------------------------------------------------------------------
__global__ void pack_rows_bf16(const float* __restrict__ src,
                               unsigned short* __restrict__ dst,
                               int R, int Tdst, int rowoff, int total) {
  int idx = blockIdx.x * blockDim.x + threadIdx.x;
  if (idx >= total) return;
  int b = idx / (R * EMB);
  int rem = idx - b * (R * EMB);
  int r = rem / EMB;
  int e = rem - r * EMB;
  dst[((size_t)b * Tdst + rowoff + r) * EMB + e] = f2bf(src[idx]);
}

// ---------------------------------------------------------------------------
// 2) Weight convert + transpose: Wt[n][k] = W[k][n] (bf16).
// ---------------------------------------------------------------------------
__global__ void transpose_cvt_bf16(const float* __restrict__ w,
                                   unsigned short* __restrict__ wt) {
  int idx = blockIdx.x * blockDim.x + threadIdx.x;
  if (idx >= EMB * EMB) return;
  int n = idx / EMB;
  int k = idx - n * EMB;
  wt[(size_t)n * EMB + k] = f2bf(w[(size_t)k * EMB + n]);
}

// ---------------------------------------------------------------------------
// 3) Key-mask additive bias: mb[b][t] = (mask==0) ? -1e30 : 0.
// ---------------------------------------------------------------------------
__global__ void build_mask_bias(const float* __restrict__ mask,
                                const float* __restrict__ cmask,
                                float* __restrict__ mb) {
  int idx = blockIdx.x * blockDim.x + threadIdx.x;
  if (idx >= BATCH * TOT) return;
  int b = idx / TOT;
  int t = idx - b * TOT;
  float m = (t < SEQ) ? mask[b * SEQ + t] : cmask[b * CTX + (t - SEQ)];
  mb[idx] = (m == 0.0f) ? -1e30f : 0.0f;
}

// ---------------------------------------------------------------------------
// 4) WMMA bf16 GEMM: C[M,N] = A[M,K] * Bt[N,K]^T. 128 threads (4 waves),
//    tile 128x128, BK=64 (two 16x16x32 k-halves -> 32 v_wmma per stage).
// ---------------------------------------------------------------------------
template <bool FINAL>
__global__ __launch_bounds__(128, 1) void gemm_wmma_bf16(
    const unsigned short* __restrict__ A,
    const unsigned short* __restrict__ Bt,
    void* __restrict__ Cout,
    const float* __restrict__ bias,
    int M, int N, int K) {
  __shared__ __align__(16) unsigned short As[128][64];
  __shared__ __align__(16) unsigned short Bs[128][64];

  const int tid  = threadIdx.x;
  const int wave = tid >> 5;
  const int lane = tid & 31;
  const int hf   = lane >> 4;
  const int l15  = lane & 15;
  const int waveM = (wave >> 1) * 64;
  const int waveN = (wave & 1) * 64;
  const int blockM = blockIdx.y * 128;
  const int blockN = blockIdx.x * 128;

  const unsigned short* aPtr = A + (size_t)(blockM + tid) * K;
  const unsigned short* bPtr = Bt + (size_t)(blockN + tid) * K;

  v8f acc[4][4] = {};

  for (int k0 = 0; k0 < K; k0 += 64) {
    __syncthreads();
    {
      const uint4* sa = (const uint4*)(aPtr + k0);
      uint4* da = (uint4*)&As[tid][0];
#pragma unroll
      for (int i = 0; i < 8; i++) da[i] = sa[i];
      const uint4* sb = (const uint4*)(bPtr + k0);
      uint4* db = (uint4*)&Bs[tid][0];
#pragma unroll
      for (int i = 0; i < 8; i++) db[i] = sb[i];
    }
    __syncthreads();

#pragma unroll
    for (int kh = 0; kh < 2; kh++) {
      FragBF16 a[4], b[4];
#pragma unroll
      for (int i = 0; i < 4; i++) {
        int m = waveM + i * 16 + l15;
        a[i].u[0] = *(const uint4*)&As[m][kh * 32 + hf * 8];
        a[i].u[1] = *(const uint4*)&As[m][kh * 32 + 16 + hf * 8];
      }
#pragma unroll
      for (int j = 0; j < 4; j++) {
        int n = waveN + j * 16 + l15;
        b[j].u[0] = *(const uint4*)&Bs[n][kh * 32 + hf * 8];
        b[j].u[1] = *(const uint4*)&Bs[n][kh * 32 + 16 + hf * 8];
      }
#pragma unroll
      for (int i = 0; i < 4; i++)
#pragma unroll
        for (int j = 0; j < 4; j++)
          acc[i][j] = __builtin_amdgcn_wmma_f32_16x16x32_bf16(
              false, a[i].v, false, b[j].v, (short)0, acc[i][j], false, false);
    }
  }

  // Epilogue: D layout = lane holds N=l15, VGPR r holds row 8*hf + r.
#pragma unroll
  for (int i = 0; i < 4; i++)
#pragma unroll
    for (int j = 0; j < 4; j++)
#pragma unroll
      for (int r = 0; r < 8; r++) {
        int row = blockM + waveM + i * 16 + hf * 8 + r;
        int col = blockN + waveN + j * 16 + l15;
        float v = acc[i][j][r];
        if (FINAL) {
          ((float*)Cout)[(size_t)row * N + col] = v + bias[col];
        } else {
          ((unsigned short*)Cout)[(size_t)row * N + col] = f2bf(v);
        }
      }
}

// ---------------------------------------------------------------------------
// 5) Per-head LayerNorm (in place, bf16) with folded 1/E^0.25 scale.
// ---------------------------------------------------------------------------
__global__ void ln_head_kernel(unsigned short* __restrict__ q,
                               const float* __restrict__ w,
                               const float* __restrict__ b,
                               int nrows) {
  int row  = blockIdx.x * 8 + (threadIdx.x >> 5);
  int lane = threadIdx.x & 31;
  if (row >= nrows) return;
  unsigned short* p = q + (size_t)row * HEADSIZE;
  float v0 = bf2f(p[lane * 2]);
  float v1 = bf2f(p[lane * 2 + 1]);

  float s = v0 + v1;
  for (int o = 16; o >= 1; o >>= 1) s += __shfl_xor(s, o, 32);
  float mu = s * (1.0f / 64.0f);
  float d0 = v0 - mu, d1 = v1 - mu;
  float sq = d0 * d0 + d1 * d1;
  for (int o = 16; o >= 1; o >>= 1) sq += __shfl_xor(sq, o, 32);
  float rstd = rsqrtf(sq * (1.0f / 64.0f) + 1e-5f);

  float o0 = (d0 * rstd * w[lane * 2]     + b[lane * 2])     * INV_SCALE;
  float o1 = (d1 * rstd * w[lane * 2 + 1] + b[lane * 2 + 1]) * INV_SCALE;
  p[lane * 2]     = f2bf(o0);
  p[lane * 2 + 1] = f2bf(o1);
}

// ---------------------------------------------------------------------------
// 6) Flash attention per (batch, head, 64-query tile). 4 waves x 16 rows.
//    K tile staged by the Tensor Data Mover (wave 0 issues one descriptor,
//    TENSORcnt-waited, barrier publishes); V tile staged transposed manually.
// ---------------------------------------------------------------------------
__global__ __launch_bounds__(128, 1) void attn_kernel(
    const unsigned short* __restrict__ Q,
    const unsigned short* __restrict__ Kb,
    const unsigned short* __restrict__ Vb,
    const float* __restrict__ mask,
    const float* __restrict__ mbias,
    unsigned short* __restrict__ O) {
  __shared__ __align__(16) unsigned short Ks[64][64];      // [key][dim]
  __shared__ __align__(16) unsigned short Vs[64][80];      // [dim][key], padded
  __shared__ __align__(16) unsigned short Ps[4][16][64];   // per-wave P tile

  const int b  = blockIdx.z;
  const int h  = blockIdx.y;
  const int q0 = blockIdx.x * 64;
  const int tid  = threadIdx.x;
  const int wave = tid >> 5;
  const int lane = tid & 31;
  const int hf   = lane >> 4;
  const int l15  = lane & 15;

  // Resident Q fragments for this wave's 16 rows (two K-halves of dim 64).
  FragBF16 aQ[2];
  {
    int s = q0 + wave * 16 + l15;
    const unsigned short* qp = Q + ((size_t)(b * SEQ + s)) * EMB + h * HEADSIZE;
#pragma unroll
    for (int kh = 0; kh < 2; kh++) {
      aQ[kh].u[0] = *(const uint4*)(qp + kh * 32 + hf * 8);
      aQ[kh].u[1] = *(const uint4*)(qp + kh * 32 + 16 + hf * 8);
    }
  }

  v8f oacc[4] = {};
  float mrow[8], lrow[8];
#pragma unroll
  for (int r = 0; r < 8; r++) { mrow[r] = -1e30f; lrow[r] = 0.0f; }

  // Incremented base pointers (avoid per-iteration 64-bit address rebuilds).
  const size_t stageOff = (size_t)(tid >> 1) * EMB + (tid & 1) * 32;
  const unsigned short* kTile = Kb + (size_t)b * TOT * EMB + h * HEADSIZE; // uniform
  const unsigned short* kPtr  = kTile + stageOff;                          // per-thread
  const unsigned short* vPtr  = Vb + (size_t)b * TOT * EMB + h * HEADSIZE + stageOff;
  const float* mbp = mbias + b * TOT + l15;

  for (int kt = 0; kt < TOT; kt += 64) {
    __syncthreads();
    // ---- Stage K tile [key][dim] ----
#if HAS_TDM
    if (wave == 0) {
      tdm_load_tile_2d((unsigned)(size_t)&Ks[0][0],
                       (unsigned long long)(size_t)kTile,
                       /*tile_d0=*/64, /*tile_d1=*/64, /*stride0=*/EMB);
    }
    kTile += (size_t)64 * EMB;
#else
    {
      const int row = tid >> 1, seg = tid & 1;
      const uint4* kp4 = (const uint4*)kPtr;
      uint4 kq[4];
#pragma unroll
      for (int i = 0; i < 4; i++) kq[i] = kp4[i];
      uint4* kd = (uint4*)&Ks[row][seg * 32];
#pragma unroll
      for (int i = 0; i < 4; i++) kd[i] = kq[i];
      if (kt + 64 < TOT)
        __builtin_prefetch((const void*)(kPtr + 64 * EMB), 0, 1);
      kPtr += (size_t)64 * EMB;
    }
#endif
    // ---- Stage V tile transposed [dim][key] (all 128 threads) ----
    {
      const int row = tid >> 1, seg = tid & 1;
      const uint4* vp4 = (const uint4*)vPtr;
      uint4 vq[4];
#pragma unroll
      for (int i = 0; i < 4; i++) vq[i] = vp4[i];
      if (kt + 64 < TOT)
        __builtin_prefetch((const void*)(vPtr + 64 * EMB), 0, 1);
      const unsigned short* tv = (const unsigned short*)vq;
#pragma unroll
      for (int d2 = 0; d2 < 32; d2++) Vs[seg * 32 + d2][row] = tv[d2];
      vPtr += (size_t)64 * EMB;
    }
#if HAS_TDM
    if (wave == 0) __builtin_amdgcn_s_wait_tensorcnt((short)0);
#endif
    __syncthreads();

    // Key-mask additive bias for this tile (same for all rows of a column).
    float kb[4];
#pragma unroll
    for (int ns = 0; ns < 4; ns++) kb[ns] = mbp[kt + ns * 16];

    // Scores: S = Q * K^T + bias, 16x64 per wave (8 v_wmma).
    v8f sc[4];
#pragma unroll
    for (int ns = 0; ns < 4; ns++) {
      FragBF16 bK[2];
      int key = ns * 16 + l15;
#pragma unroll
      for (int kh = 0; kh < 2; kh++) {
        bK[kh].u[0] = *(const uint4*)&Ks[key][kh * 32 + hf * 8];
        bK[kh].u[1] = *(const uint4*)&Ks[key][kh * 32 + 16 + hf * 8];
      }
      v8f z = {};
      z = __builtin_amdgcn_wmma_f32_16x16x32_bf16(false, aQ[0].v, false, bK[0].v,
                                                  (short)0, z, false, false);
      z = __builtin_amdgcn_wmma_f32_16x16x32_bf16(false, aQ[1].v, false, bK[1].v,
                                                  (short)0, z, false, false);
#pragma unroll
      for (int r = 0; r < 8; r++) z[r] += kb[ns];
      sc[ns] = z;
    }

    // Online softmax per row (row = 8*hf + r); P -> per-wave LDS immediately.
#pragma unroll
    for (int r = 0; r < 8; r++) {
      float mx = fmaxf(fmaxf(sc[0][r], sc[1][r]), fmaxf(sc[2][r], sc[3][r]));
      mx = redmax16(mx);
      float mnew = fmaxf(mrow[r], mx);
      float rescale = __expf(mrow[r] - mnew);
      float rs = 0.0f;
      unsigned short* prow = &Ps[wave][hf * 8 + r][l15];
#pragma unroll
      for (int ns = 0; ns < 4; ns++) {
        float p = __expf(sc[ns][r] - mnew);   // masked: underflows to 0
        rs += p;
        prow[ns * 16] = f2bf(p);
      }
      rs = redsum16(rs);
      lrow[r] = lrow[r] * rescale + rs;
      mrow[r] = mnew;
#pragma unroll
      for (int ds = 0; ds < 4; ds++) oacc[ds][r] *= rescale;
    }

    // P @ V (LDS ops are in-order within a wave; Ps region is wave-private).
    FragBF16 aP[2];
#pragma unroll
    for (int kh = 0; kh < 2; kh++) {
      aP[kh].u[0] = *(const uint4*)&Ps[wave][l15][kh * 32 + hf * 8];
      aP[kh].u[1] = *(const uint4*)&Ps[wave][l15][kh * 32 + 16 + hf * 8];
    }
#pragma unroll
    for (int ds = 0; ds < 4; ds++) {
      FragBF16 bV[2];
      int d = ds * 16 + l15;
#pragma unroll
      for (int kh = 0; kh < 2; kh++) {
        bV[kh].u[0] = *(const uint4*)&Vs[d][kh * 32 + hf * 8];
        bV[kh].u[1] = *(const uint4*)&Vs[d][kh * 32 + 16 + hf * 8];
      }
      oacc[ds] = __builtin_amdgcn_wmma_f32_16x16x32_bf16(false, aP[0].v, false, bV[0].v,
                                                         (short)0, oacc[ds], false, false);
      oacc[ds] = __builtin_amdgcn_wmma_f32_16x16x32_bf16(false, aP[1].v, false, bV[1].v,
                                                         (short)0, oacc[ds], false, false);
    }
  }

  // Epilogue: O = qmask * acc / (l + 1e-13), bf16.
  unsigned short* oBase =
      O + ((size_t)(b * SEQ + q0 + wave * 16 + hf * 8)) * EMB + h * HEADSIZE + l15;
  const float* qm = mask + b * SEQ + q0 + wave * 16 + hf * 8;
#pragma unroll
  for (int r = 0; r < 8; r++) {
    float inv = qm[r] / (lrow[r] + 1e-13f);
#pragma unroll
    for (int ds = 0; ds < 4; ds++)
      oBase[(size_t)r * EMB + ds * 16] = f2bf(oacc[ds][r] * inv);
  }
}

// ---------------------------------------------------------------------------
// Host-side orchestration (all launches on `stream`; no sync/alloc).
// Workspace layout (~80.1 MB):
//   [ 0MB) x_bf16 8MB  [ 8MB) xc_bf16 16MB  [24MB) WqT [26) WkT [28) WvT [30) WuT
//   [32MB) Q 8MB  [40MB) K 16MB  [56MB) V 16MB  [72MB) attnOut 8MB  [80MB) maskBias
// ---------------------------------------------------------------------------
extern "C" void kernel_launch(void* const* d_in, const int* in_sizes, int n_in,
                              void* d_out, int out_size, void* d_ws, size_t ws_size,
                              hipStream_t stream) {
  const float* x     = (const float*)d_in[0];
  const float* ctx   = (const float*)d_in[1];
  const float* maskp = (const float*)d_in[2];
  const float* cmask = (const float*)d_in[3];
  const float* Wq    = (const float*)d_in[4];
  const float* Wk    = (const float*)d_in[5];
  const float* Wv    = (const float*)d_in[6];
  const float* Wu    = (const float*)d_in[7];
  const float* bu    = (const float*)d_in[8];
  const float* qlnw  = (const float*)d_in[9];
  const float* qlnb  = (const float*)d_in[10];
  const float* klnw  = (const float*)d_in[11];
  const float* klnb  = (const float*)d_in[12];

  char* ws = (char*)d_ws;
  const size_t MB = 1u << 20;
  unsigned short* xb  = (unsigned short*)(ws + 0 * MB);
  unsigned short* xcb = (unsigned short*)(ws + 8 * MB);
  unsigned short* wqT = (unsigned short*)(ws + 24 * MB);
  unsigned short* wkT = (unsigned short*)(ws + 26 * MB);
  unsigned short* wvT = (unsigned short*)(ws + 28 * MB);
  unsigned short* wuT = (unsigned short*)(ws + 30 * MB);
  unsigned short* Qb  = (unsigned short*)(ws + 32 * MB);
  unsigned short* Kb  = (unsigned short*)(ws + 40 * MB);
  unsigned short* Vb  = (unsigned short*)(ws + 56 * MB);
  unsigned short* Ob  = (unsigned short*)(ws + 72 * MB);
  float*          mb  = (float*)(ws + 80 * MB);

  const int nX = BATCH * SEQ * EMB;
  const int nC = BATCH * CTX * EMB;
  const int nW = EMB * EMB;
  const int nM = BATCH * TOT;

  // bf16 conversions / concatenation / mask bias
  pack_rows_bf16<<<(nX + 255) / 256, 256, 0, stream>>>(x,   xb,  SEQ, SEQ, 0,   nX);
  pack_rows_bf16<<<(nX + 255) / 256, 256, 0, stream>>>(x,   xcb, SEQ, TOT, 0,   nX);
  pack_rows_bf16<<<(nC + 255) / 256, 256, 0, stream>>>(ctx, xcb, CTX, TOT, SEQ, nC);
  transpose_cvt_bf16<<<(nW + 255) / 256, 256, 0, stream>>>(Wq, wqT);
  transpose_cvt_bf16<<<(nW + 255) / 256, 256, 0, stream>>>(Wk, wkT);
  transpose_cvt_bf16<<<(nW + 255) / 256, 256, 0, stream>>>(Wv, wvT);
  transpose_cvt_bf16<<<(nW + 255) / 256, 256, 0, stream>>>(Wu, wuT);
  build_mask_bias<<<(nM + 255) / 256, 256, 0, stream>>>(maskp, cmask, mb);

  // QKV projections (WMMA bf16)
  gemm_wmma_bf16<false><<<dim3(EMB / 128, (BATCH * SEQ) / 128), 128, 0, stream>>>(
      xb, wqT, (void*)Qb, nullptr, BATCH * SEQ, EMB, EMB);
  gemm_wmma_bf16<false><<<dim3(EMB / 128, (BATCH * TOT) / 128), 128, 0, stream>>>(
      xcb, wkT, (void*)Kb, nullptr, BATCH * TOT, EMB, EMB);
  gemm_wmma_bf16<false><<<dim3(EMB / 128, (BATCH * TOT) / 128), 128, 0, stream>>>(
      xcb, wvT, (void*)Vb, nullptr, BATCH * TOT, EMB, EMB);

  // Per-head LayerNorm (+ 1/E^0.25 folded), in place
  const int qRows = BATCH * SEQ * HEADS;
  const int kRows = BATCH * TOT * HEADS;
  ln_head_kernel<<<qRows / 8, 256, 0, stream>>>(Qb, qlnw, qlnb, qRows);
  ln_head_kernel<<<kRows / 8, 256, 0, stream>>>(Kb, klnw, klnb, kRows);

  // Flash attention (TDM-staged K tiles)
  attn_kernel<<<dim3(SEQ / 64, HEADS, BATCH), 128, 0, stream>>>(
      Qb, Kb, Vb, maskp, mb, Ob);

  // Output projection with fused bias -> fp32 d_out
  gemm_wmma_bf16<true><<<dim3(EMB / 128, (BATCH * SEQ) / 128), 128, 0, stream>>>(
      Ob, wuT, d_out, bu, BATCH * SEQ, EMB, EMB);
}
